// RelPartialLearnableMultiHeadAttn_35613868819145
// MI455X (gfx1250) — matrix operand order repaired and checked by reading, not compiled, not run on hardware
//
#include <hip/hip_runtime.h>
#include <hip/hip_bf16.h>

// ---------------------------------------------------------------------------
// Transformer-XL RelPartialLearnableMultiHeadAttn for MI455X (gfx1250).
// All GEMMs via v_wmma_f32_16x16x32_bf16 (wave32). Flash-style attention with
// analytic rel_shift (BD[i,j] = BDraw[i, j+Q-1-i] on unmasked positions).
// GEMM stages B tiles through LDS (double buffered, shared by 4 waves),
// using GLOBAL_LOAD_ASYNC_TO_LDS_B128 (ASYNCcnt) when the toolchain exposes it.
// Workspace usage ~158 MB.
// ---------------------------------------------------------------------------

#define NHEAD  16
#define DHEAD  64
#define DMODEL 1024
#define QLEN   1024
#define MLEN   1024
#define BSZ    4
#define KLEN   (QLEN + MLEN)

typedef unsigned short u16;
typedef __attribute__((ext_vector_type(16))) __bf16       v16bf;
typedef __attribute__((ext_vector_type(8)))  float        v8f;
typedef __attribute__((ext_vector_type(4)))  unsigned int u32x4;

// CDNA5 async global->LDS DMA (ASYNCcnt-tracked), if the compiler exposes it.
#if defined(__has_builtin)
#if __has_builtin(__builtin_amdgcn_global_load_async_to_lds_b128) && \
    __has_builtin(__builtin_amdgcn_s_wait_asynccnt)
#define HAVE_ASYNC_LDS 1
#endif
#endif
#ifndef HAVE_ASYNC_LDS
#define HAVE_ASYNC_LDS 0
#endif

#if HAVE_ASYNC_LDS
// Exact parameter pointee type per clang diagnostic:
//   __attribute__((__vector_size__(4 * sizeof(int)))) int
typedef int v4i_gcc __attribute__((__vector_size__(4 * sizeof(int))));
typedef __attribute__((address_space(1))) v4i_gcc* as1_v4i;
typedef __attribute__((address_space(3))) v4i_gcc* as3_v4i;
#endif

union Frag { v16bf v; u32x4 q[2]; };

__device__ __forceinline__ u16 f2bf(float f) {
  union { float f; unsigned u; } x; x.f = f;
  unsigned u = x.u + 0x7FFFu + ((x.u >> 16) & 1u);   // round-to-nearest-even
  return (u16)(u >> 16);
}
__device__ __forceinline__ float bf2f(u16 h) {
  union { unsigned u; float f; } x; x.u = ((unsigned)h) << 16;
  return x.f;
}

// A fragment: 16x32 bf16, src points at (row0,k0) of row-major [*, ld].
// lanes 0-15: M=lane, K=0..15; lanes 16-31: M=lane-16, K=16..31
__device__ __forceinline__ v16bf ldA(const u16* src, int ld, int lane) {
  const int m = lane & 15, hi = lane >> 4;
  const u16* p = src + (size_t)m * ld + hi * 16;
  Frag f;
  f.q[0] = *(const u32x4*)(p);        // K = k0 + hi*16 .. +7
  f.q[1] = *(const u32x4*)(p + 8);    // K = k0 + hi*16 + 8 .. +15
  return f.v;
}
// B fragment: 32x16 bf16 from B^T storage (N x K row-major). srcT at (n0,k0).
// lanes 0-15: N=lane, K=0..15; lanes 16-31: N=lane-16, K=16..31
__device__ __forceinline__ v16bf ldB(const u16* srcT, int ld, int lane) {
  const int n = lane & 15, hi = lane >> 4;
  const u16* p = srcT + (size_t)n * ld + hi * 16;
  Frag f;
  f.q[0] = *(const u32x4*)(p);
  f.q[1] = *(const u32x4*)(p + 8);
  return f.v;
}
// B fragment from an LDS-staged [64][32] tile; srcT at row 16*t, k stride 32.
__device__ __forceinline__ v16bf ldBs(const u16* srcT, int lane) {
  const int n = lane & 15, hi = lane >> 4;
  const u16* p = srcT + n * 32 + hi * 16;
  Frag f;
  f.q[0] = *(const u32x4*)(p);
  f.q[1] = *(const u32x4*)(p + 8);
  return f.v;
}
// Same as ldB but clamps the column index (BD window edges; clamped lanes are
// always masked later).
__device__ __forceinline__ v16bf ldBclamp(const u16* base, int ld, int col0,
                                          int maxcol, int k0, int lane) {
  const int n = lane & 15, hi = lane >> 4;
  int c = col0 + n;
  c = c < 0 ? 0 : (c > maxcol ? maxcol : c);
  const u16* p = base + (size_t)c * ld + k0 + hi * 16;
  Frag f;
  f.q[0] = *(const u32x4*)(p);
  f.q[1] = *(const u32x4*)(p + 8);
  return f.v;
}
__device__ __forceinline__ v8f wmma_bf16(v16bf a, v16bf b, v8f c) {
  return __builtin_amdgcn_wmma_f32_16x16x32_bf16(false, a, false, b,
                                                 (short)0, c, false, false);
}

// Stage a 64x32 bf16 B tile (global, B^T layout, ld=Kd) into LDS.
// 128 threads x 2 chunks x 16B = 4 KB.
__device__ __forceinline__ void stageB(const u16* __restrict__ gsrc, int Kd,
                                       u16* lds, int tid) {
#pragma unroll
  for (int h = 0; h < 2; ++h) {
    const int chunk = tid + h * 128;          // 0..255
    const int n  = chunk >> 2;                // 0..63
    const int kc = (chunk & 3) * 8;           // 0,8,16,24
    const u16* g = gsrc + (size_t)n * Kd + kc;
    u16* l = lds + n * 32 + kc;
#if HAVE_ASYNC_LDS
    __builtin_amdgcn_global_load_async_to_lds_b128((as1_v4i)g, (as3_v4i)l, 0, 0);
#else
    *(u32x4*)l = *(const u32x4*)g;
#endif
  }
}

// ------------------------------- converters --------------------------------
__global__ void k_cat(const float* __restrict__ mems, const float* __restrict__ w,
                      u16* __restrict__ dst, long n) {
  long i = (long)blockIdx.x * blockDim.x + threadIdx.x;
  if (i >= n) return;
  const long half = (long)MLEN * BSZ * DMODEL;
  float v = (i < half) ? mems[i] : w[i - half];
  dst[i] = f2bf(v);
}
__global__ void k_conv(const float* __restrict__ src, u16* __restrict__ dst, long n) {
  long i = (long)blockIdx.x * blockDim.x + threadIdx.x;
  if (i < n) dst[i] = f2bf(src[i]);
}
__global__ void k_tconv(const float* __restrict__ src, u16* __restrict__ dst,
                        int rows, int cols) {  // dst[c][r] = src[r][c]
  long i = (long)blockIdx.x * blockDim.x + threadIdx.x;
  long n = (long)rows * cols;
  if (i >= n) return;
  int rr = (int)(i / cols), cc = (int)(i % cols);
  dst[(long)cc * rows + rr] = f2bf(src[i]);
}

// ------------------------- generic bf16 WMMA GEMM --------------------------
// C[M,N] (bf16) = A[M,K] (bf16, row-major) @ B (given as B^T: [N,K] row-major)
// Workgroup = 4 waves -> 64x64 C tile; B tiles double-buffered through LDS.
__global__ __launch_bounds__(128)
void k_gemm(const u16* __restrict__ A, const u16* __restrict__ Bt,
            u16* __restrict__ C, int M, int N, int Kd) {
  __shared__ __align__(16) u16 sB[2][64][32];
  const int tid  = threadIdx.x;
  const int lane = tid & 31;
  const int wv   = tid >> 5;
  const int r0   = blockIdx.y * 64 + wv * 16;
  const int c0   = blockIdx.x * 64;
  const v8f z = {0.f,0.f,0.f,0.f,0.f,0.f,0.f,0.f};
  v8f a0 = z, a1 = z, a2 = z, a3 = z;
  const u16* Ar = A  + (size_t)r0 * Kd;
  const u16* Bc = Bt + (size_t)c0 * Kd;

  stageB(Bc, Kd, &sB[0][0][0], tid);
  for (int k0 = 0; k0 < Kd; k0 += 32) {
    const int buf = (k0 >> 5) & 1;
#if HAVE_ASYNC_LDS
    __builtin_amdgcn_s_wait_asynccnt(0);
#endif
    __syncthreads();                                   // staged tile visible
    if (k0 + 32 < Kd) stageB(Bc + (k0 + 32), Kd, &sB[buf ^ 1][0][0], tid);
    __builtin_prefetch(Ar + k0 + 64, 0, 1);            // next-next A chunk
    v16bf a = ldA(Ar + k0, Kd, lane);
    const u16* sb = &sB[buf][0][0];
    v16bf b0 = ldBs(sb +  0 * 32, lane);
    v16bf b1 = ldBs(sb + 16 * 32, lane);
    v16bf b2 = ldBs(sb + 32 * 32, lane);
    v16bf b3 = ldBs(sb + 48 * 32, lane);
    a0 = wmma_bf16(a, b0, a0);
    a1 = wmma_bf16(a, b1, a1);
    a2 = wmma_bf16(a, b2, a2);
    a3 = wmma_bf16(a, b3, a3);
  }
  const int hi = lane >> 4, nn = lane & 15;
#pragma unroll
  for (int v = 0; v < 8; ++v) {
    size_t row = (size_t)(r0 + v + 8 * hi) * N;
    C[row + c0 +  0 + nn] = f2bf(a0[v]);
    C[row + c0 + 16 + nn] = f2bf(a1[v]);
    C[row + c0 + 32 + nn] = f2bf(a2[v]);
    C[row + c0 + 48 + nn] = f2bf(a3[v]);
  }
}

// ------------------------------ repack stages ------------------------------
// heads[(t*BSZ+b)][3072]: [0,1024)=q, [1024,2048)=k, [2048,3072)=v
__global__ void k_repack_qkv(const u16* __restrict__ heads,
                             const float* __restrict__ rwb, const float* __restrict__ rrb,
                             u16* __restrict__ qrw, u16* __restrict__ qrr,
                             u16* __restrict__ kk, u16* __restrict__ vT) {
  long idx = (long)blockIdx.x * blockDim.x + threadIdx.x;
  const long tot = (long)KLEN * BSZ * NHEAD * DHEAD;
  if (idx >= tot) return;
  const int  d = (int)(idx & 63);
  const int  n = (int)((idx >> 6) & 15);
  const int  b = (int)((idx >> 10) & 3);
  const long t = idx >> 12;
  const size_t row = (size_t)t * BSZ + b;
  const int    nd  = n * 64 + d;
  const size_t bn  = (size_t)b * NHEAD + n;
  kk[(bn * KLEN + t) * DHEAD + d] = heads[row * 3072 + 1024 + nd];
  vT[(bn * DHEAD + d) * KLEN + t] = heads[row * 3072 + 2048 + nd];
  if (t >= MLEN) {
    const long i = t - MLEN;
    const float q = bf2f(heads[row * 3072 + nd]);
    qrw[(bn * QLEN + i) * DHEAD + d] = f2bf(q + rwb[nd]);
    qrr[(bn * QLEN + i) * DHEAD + d] = f2bf(q + rrb[nd]);
  }
}
// rk[n][jr][d] = rkraw[jr][n*64+d]
__global__ void k_repack_rk(const u16* __restrict__ rkraw, u16* __restrict__ rk) {
  long idx = (long)blockIdx.x * blockDim.x + threadIdx.x;
  const long tot = (long)NHEAD * KLEN * DHEAD;
  if (idx >= tot) return;
  const int  d  = (int)(idx & 63);
  const long jr = (idx >> 6) & (KLEN - 1);
  const int  n  = (int)(idx >> 17);
  rk[((size_t)n * KLEN + jr) * DHEAD + d] = rkraw[jr * DMODEL + n * 64 + d];
}

// ------------------------------ attention ----------------------------------
// grid (QLEN/64, NHEAD, BSZ), 4 waves; wave owns 16 Q rows, flash loop over j.
__global__ __launch_bounds__(128)
void k_attn(const u16* __restrict__ qrw, const u16* __restrict__ qrr,
            const u16* __restrict__ kk, const u16* __restrict__ vT,
            const u16* __restrict__ rk, u16* __restrict__ vec) {
  __shared__ float sbd[4][16][48];                 // BDraw window per wave
  __shared__ __align__(16) u16 sp[4][16][32];      // P tile (C->A relayout)
  const int lane = threadIdx.x & 31;
  const int wv   = threadIdx.x >> 5;
  const int b    = blockIdx.z, nh = blockIdx.y;
  const int i0   = blockIdx.x * 64 + wv * 16;
  const size_t bn = (size_t)b * NHEAD + nh;
  const u16* Qw = qrw + (bn * QLEN + i0) * DHEAD;
  const u16* Qr = qrr + (bn * QLEN + i0) * DHEAD;
  const u16* Kp = kk + bn * KLEN * DHEAD;
  const u16* Vt = vT + bn * (size_t)DHEAD * KLEN;
  const u16* Rk = rk + (size_t)nh * KLEN * DHEAD;

  const v16bf aW0 = ldA(Qw,      DHEAD, lane);     // Q+r_w_bias, K(d)=0..31
  const v16bf aW1 = ldA(Qw + 32, DHEAD, lane);     // K(d)=32..63
  const v16bf aR0 = ldA(Qr,      DHEAD, lane);
  const v16bf aR1 = ldA(Qr + 32, DHEAD, lane);

  const v8f z = {0.f,0.f,0.f,0.f,0.f,0.f,0.f,0.f};
  v8f o0 = z, o1 = z, o2 = z, o3 = z;              // O tile 16x64
  float mi[8], li[8];
#pragma unroll
  for (int v = 0; v < 8; ++v) { mi[v] = -3.0e38f; li[v] = 0.0f; }

  const int hi = lane >> 4, nn = lane & 15;
  const float scale = 0.125f;                      // 1/sqrt(64)
  const int jmax = MLEN + i0 + 15;                 // last unmasked column
  for (int j0 = 0; j0 <= jmax; j0 += 32) {
    if (j0 + 32 <= jmax) {                         // hint next streamed tiles
      __builtin_prefetch(Kp + (size_t)(j0 + 32) * DHEAD, 0, 1);
      __builtin_prefetch(Vt + (size_t)(j0 + 32), 0, 1);
    }
    // ---- AC = (q+rw) @ K^T, 16x32 ----
    v8f ac0 = z, ac1 = z;
    ac0 = wmma_bf16(aW0, ldB(Kp + (size_t)j0 * DHEAD,             DHEAD, lane), ac0);
    ac0 = wmma_bf16(aW1, ldB(Kp + (size_t)j0 * DHEAD + 32,        DHEAD, lane), ac0);
    ac1 = wmma_bf16(aW0, ldB(Kp + (size_t)(j0 + 16) * DHEAD,      DHEAD, lane), ac1);
    ac1 = wmma_bf16(aW1, ldB(Kp + (size_t)(j0 + 16) * DHEAD + 32, DHEAD, lane), ac1);

    // ---- BDraw window (16x48) -> LDS; BD[i,j] = BDraw[i, j+QLEN-1-i] ----
    const int jrb = j0 + QLEN - 16 - i0;
#pragma unroll
    for (int t = 0; t < 3; ++t) {
      v8f bd = z;
      bd = wmma_bf16(aR0, ldBclamp(Rk, DHEAD, jrb + 16 * t, KLEN - 1, 0,  lane), bd);
      bd = wmma_bf16(aR1, ldBclamp(Rk, DHEAD, jrb + 16 * t, KLEN - 1, 32, lane), bd);
#pragma unroll
      for (int v = 0; v < 8; ++v) sbd[wv][v + 8 * hi][16 * t + nn] = bd[v];
    }
    __threadfence_block();

    // ---- S = scale*(AC + shifted BD), mask, online softmax ----
#pragma unroll
    for (int v = 0; v < 8; ++v) {
      const int m  = v + 8 * hi;
      const int ig = i0 + m;
      float x0 = (ac0[v] + sbd[wv][m][nn + 15 - m]) * scale;
      float x1 = (ac1[v] + sbd[wv][m][nn + 31 - m]) * scale;
      if (j0 + nn      > MLEN + ig) x0 = -1.0e30f;
      if (j0 + 16 + nn > MLEN + ig) x1 = -1.0e30f;
      float tmax = fmaxf(x0, x1);
#pragma unroll
      for (int d = 1; d < 16; d <<= 1) tmax = fmaxf(tmax, __shfl_xor(tmax, d, 32));
      const float mnew  = fmaxf(mi[v], tmax);
      const float alpha = __expf(mi[v] - mnew);
      const float p0 = __expf(x0 - mnew);
      const float p1 = __expf(x1 - mnew);
      float rs = p0 + p1;
#pragma unroll
      for (int d = 1; d < 16; d <<= 1) rs += __shfl_xor(rs, d, 32);
      li[v] = li[v] * alpha + rs;
      mi[v] = mnew;
      o0[v] = o0[v] * alpha; o1[v] = o1[v] * alpha;
      o2[v] = o2[v] * alpha; o3[v] = o3[v] * alpha;
      sp[wv][m][nn]      = f2bf(p0);
      sp[wv][m][16 + nn] = f2bf(p1);
    }
    __threadfence_block();

    // ---- O += P @ V (P re-read from LDS in A-fragment layout) ----
    v16bf aP;
    {
      const u16* p = &sp[wv][nn][hi * 16];
      Frag f;
      f.q[0] = *(const u32x4*)(p);
      f.q[1] = *(const u32x4*)(p + 8);
      aP = f.v;
    }
    o0 = wmma_bf16(aP, ldB(Vt + (size_t) 0 * KLEN + j0, KLEN, lane), o0);
    o1 = wmma_bf16(aP, ldB(Vt + (size_t)16 * KLEN + j0, KLEN, lane), o1);
    o2 = wmma_bf16(aP, ldB(Vt + (size_t)32 * KLEN + j0, KLEN, lane), o2);
    o3 = wmma_bf16(aP, ldB(Vt + (size_t)48 * KLEN + j0, KLEN, lane), o3);
  }

#pragma unroll
  for (int v = 0; v < 8; ++v) {
    const float inv = 1.0f / li[v];
    const int ig = i0 + v + 8 * hi;
    const size_t base = ((size_t)ig * BSZ + b) * DMODEL + nh * DHEAD;
    vec[base +  0 + nn] = f2bf(o0[v] * inv);
    vec[base + 16 + nn] = f2bf(o1[v] * inv);
    vec[base + 32 + nn] = f2bf(o2[v] * inv);
    vec[base + 48 + nn] = f2bf(o3[v] * inv);
  }
}

// -------------------------- residual + LayerNorm ---------------------------
__global__ __launch_bounds__(256)
void k_ln(const float* __restrict__ w, const u16* __restrict__ attn,
          const float* __restrict__ gamma, const float* __restrict__ beta,
          float* __restrict__ out) {
  __shared__ float red0[8], red1[8];
  const int row = blockIdx.x;
  const size_t base = (size_t)row * DMODEL;
  float x[4], sum = 0.f, sumsq = 0.f;
#pragma unroll
  for (int t = 0; t < 4; ++t) {
    const int c = threadIdx.x + t * 256;
    const float v = w[base + c] + bf2f(attn[base + c]);
    x[t] = v; sum += v; sumsq += v * v;
  }
#pragma unroll
  for (int d = 1; d < 32; d <<= 1) {
    sum   += __shfl_xor(sum,   d, 32);
    sumsq += __shfl_xor(sumsq, d, 32);
  }
  const int wv = threadIdx.x >> 5, ln = threadIdx.x & 31;
  if (ln == 0) { red0[wv] = sum; red1[wv] = sumsq; }
  __syncthreads();
  if (wv == 0) {
    float s = (ln < 8) ? red0[ln] : 0.f;
    float q = (ln < 8) ? red1[ln] : 0.f;
#pragma unroll
    for (int d = 1; d < 8; d <<= 1) { s += __shfl_xor(s, d, 32); q += __shfl_xor(q, d, 32); }
    if (ln == 0) { red0[0] = s; red1[0] = q; }
  }
  __syncthreads();
  const float mean = red0[0] * (1.0f / DMODEL);
  const float var  = red1[0] * (1.0f / DMODEL) - mean * mean;
  const float rstd = rsqrtf(var + 1e-5f);
#pragma unroll
  for (int t = 0; t < 4; ++t) {
    const int c = threadIdx.x + t * 256;
    out[base + c] = (x[t] - mean) * rstd * gamma[c] + beta[c];
  }
}

// --------------------------------- driver ----------------------------------
extern "C" void kernel_launch(void* const* d_in, const int* in_sizes, int n_in,
                              void* d_out, int out_size, void* d_ws, size_t ws_size,
                              hipStream_t stream) {
  (void)in_sizes; (void)n_in; (void)out_size; (void)ws_size;
  const float* w    = (const float*)d_in[0];
  const float* r    = (const float*)d_in[1];
  const float* mems = (const float*)d_in[2];
  /* d_in[3] attn_mask: computed analytically, unused */
  const float* Wqkv = (const float*)d_in[4];
  const float* Wr   = (const float*)d_in[5];
  const float* Wo   = (const float*)d_in[6];
  const float* rwb  = (const float*)d_in[7];
  const float* rrb  = (const float*)d_in[8];
  const float* gam  = (const float*)d_in[9];
  const float* bet  = (const float*)d_in[10];
  float* out = (float*)d_out;

  char* ws = (char*)d_ws;
  size_t off = 0;
  auto alloc = [&](size_t elems) {
    char* p = ws + off;
    off += (elems * sizeof(u16) + 255) & ~(size_t)255;
    return (u16*)p;
  };
  u16* catb  = alloc((size_t)KLEN * BSZ * DMODEL);        // 16.8 MB
  u16* wqkvT = alloc((size_t)3 * DMODEL * DMODEL);        //  6.3 MB
  u16* wrT   = alloc((size_t)DMODEL * DMODEL);            //  2.1 MB
  u16* woT   = alloc((size_t)DMODEL * DMODEL);            //  2.1 MB
  u16* rb    = alloc((size_t)KLEN * DMODEL);              //  4.2 MB
  u16* heads = alloc((size_t)KLEN * BSZ * 3 * DMODEL);    // 50.3 MB
  u16* rkraw = alloc((size_t)KLEN * DMODEL);              //  4.2 MB
  u16* qrw   = alloc((size_t)BSZ * NHEAD * QLEN * DHEAD); //  8.4 MB
  u16* qrr   = alloc((size_t)BSZ * NHEAD * QLEN * DHEAD); //  8.4 MB
  u16* kk    = alloc((size_t)BSZ * NHEAD * KLEN * DHEAD); // 16.8 MB
  u16* vT    = alloc((size_t)BSZ * NHEAD * KLEN * DHEAD); // 16.8 MB
  u16* rkb   = alloc((size_t)NHEAD * KLEN * DHEAD);       //  4.2 MB
  u16* vec   = alloc((size_t)QLEN * BSZ * DMODEL);        //  8.4 MB
  u16* attnb = alloc((size_t)QLEN * BSZ * DMODEL);        //  8.4 MB

  // stage 0: bf16 conversion / weight transposes
  { long n = (long)KLEN * BSZ * DMODEL;
    k_cat<<<dim3((unsigned)(n / 256)), 256, 0, stream>>>(mems, w, catb, n); }
  { long n = (long)DMODEL * 3 * DMODEL;
    k_tconv<<<dim3((unsigned)(n / 256)), 256, 0, stream>>>(Wqkv, wqkvT, DMODEL, 3 * DMODEL); }
  { long n = (long)DMODEL * DMODEL;
    k_tconv<<<dim3((unsigned)(n / 256)), 256, 0, stream>>>(Wr, wrT, DMODEL, DMODEL); }
  { long n = (long)DMODEL * DMODEL;
    k_tconv<<<dim3((unsigned)(n / 256)), 256, 0, stream>>>(Wo, woT, DMODEL, DMODEL); }
  { long n = (long)KLEN * DMODEL;
    k_conv<<<dim3((unsigned)(n / 256)), 256, 0, stream>>>(r, rb, n); }

  // stage 1: heads = cat @ Wqkv ; rkraw = r @ Wr
  { dim3 g(3 * DMODEL / 64, KLEN * BSZ / 64);
    k_gemm<<<g, 128, 0, stream>>>(catb, wqkvT, heads, KLEN * BSZ, 3 * DMODEL, DMODEL); }
  { dim3 g(DMODEL / 64, KLEN / 64);
    k_gemm<<<g, 128, 0, stream>>>(rb, wrT, rkraw, KLEN, DMODEL, DMODEL); }

  // stage 2: repack q/k/v + rel keys
  { long n = (long)KLEN * BSZ * NHEAD * DHEAD;
    k_repack_qkv<<<dim3((unsigned)(n / 256)), 256, 0, stream>>>(heads, rwb, rrb, qrw, qrr, kk, vT); }
  { long n = (long)NHEAD * KLEN * DHEAD;
    k_repack_rk<<<dim3((unsigned)(n / 256)), 256, 0, stream>>>(rkraw, rkb); }

  // stage 3: fused flash attention with rel-shift
  { dim3 g(QLEN / 64, NHEAD, BSZ);
    k_attn<<<g, 128, 0, stream>>>(qrw, qrr, kk, vT, rkb, vec); }

  // stage 4: attn_out = vec @ Wo
  { dim3 g(DMODEL / 64, QLEN * BSZ / 64);
    k_gemm<<<g, 128, 0, stream>>>(vec, woT, attnb, QLEN * BSZ, DMODEL, DMODEL); }

  // stage 5: residual + LayerNorm (f32)
  k_ln<<<dim3(QLEN * BSZ), 256, 0, stream>>>(w, attnb, gam, bet, out);
}
